// BARefiner_47837345743305
// MI455X (gfx1250) — compile-verified, matrix operand order checked by reading, not run on hardware
//
#include <hip/hip_runtime.h>
#include <hip/hip_bf16.h>

// ---------------------------------------------------------------------------
// BARefiner on MI455X (gfx1250, wave32, WMMA).
// Only params["layers"][2] affects the output (the reference loop re-feeds the
// ORIGINAL pooled_features and discards intermediate `out`), so we compute
// layer 2 + heads only. All dense GEMMs run on v_wmma_f32_16x16x32_bf16
// (bf16 in, f32 accum). Each wave computes a 16x64 output strip; per K-step
// all 4 B tiles are loaded into distinct registers BEFORE the 4 WMMAs so the
// compiler can issue partial s_wait_loadcnt and overlap loads with WMMAs.
// Segment-max: -inf init + float atomic max + PyG empty->0 fixup.
// ---------------------------------------------------------------------------

#define BN 8192
#define SD 256
#define NE 131072          // edges
#define EPB 64             // edges per block in the edge kernel

typedef __bf16 v16bf __attribute__((ext_vector_type(16)));
typedef float  v8f   __attribute__((ext_vector_type(8)));

union ABpack { uint4 q[2]; v16bf v; };

__device__ __forceinline__ unsigned short f2bf(float f) {
    unsigned u = __float_as_uint(f);
    unsigned r = u + 0x7FFFu + ((u >> 16) & 1u);   // round-to-nearest-even
    return (unsigned short)(r >> 16);
}
__device__ __forceinline__ float bf2f(unsigned short h) {
    return __uint_as_float(((unsigned)h) << 16);
}
__device__ __forceinline__ void atomic_max_f32(float* p, float v) {
#if defined(__has_builtin) && __has_builtin(__hip_atomic_fetch_max)
    __hip_atomic_fetch_max(p, v, __ATOMIC_RELAXED, __HIP_MEMORY_SCOPE_AGENT);
#else
    atomicMax(p, v);
#endif
}

// --- f32 -> bf16 arena conversion (handles submatrix extraction) -----------
__global__ __launch_bounds__(256)
void cvt_k(const float* __restrict__ src, int srcStride, int colOff,
           unsigned short* __restrict__ dst, int rows, int cols) {
    int t = blockIdx.x * 256 + threadIdx.x;
    if (t >= rows * cols) return;
    int r = t / cols, c = t % cols;
    dst[t] = f2bf(src[(size_t)r * srcStride + colOff + c]);
}

__global__ __launch_bounds__(256)
void fill_neginf_k(float* __restrict__ p, int n) {
    int t = blockIdx.x * 256 + threadIdx.x;
    if (t < n) p[t] = -INFINITY;
}

__global__ __launch_bounds__(256)
void fix_agg_k(const float* __restrict__ agg, unsigned short* __restrict__ aggb, int n) {
    int t = blockIdx.x * 256 + threadIdx.x;
    if (t >= n) return;
    float v = agg[t];
    if (isinf(v) && v < 0.f) v = 0.f;               // PyG: empty segment -> 0
    aggb[t] = f2bf(v);
}

// --- generic WMMA GEMM: C[M,N] = act(A[M,K] @ W[N,K]^T + bias (+resid)) ----
// A, W stored bf16 row-major. One 16(M) x 64(N) strip per wave.
// 4 waves/block. Requires N % 64 == 0, K % 32 == 0.
__global__ __launch_bounds__(128)
void gemm_bf16_k(const unsigned short* __restrict__ A,
                 const unsigned short* __restrict__ Bw,
                 const float* __restrict__ bias,
                 const float* __restrict__ resid,
                 float* __restrict__ Cf, unsigned short* __restrict__ Cb,
                 int M, int N, int K, int relu) {
    int ngroups = N >> 6;                            // N / 64
    int tile    = blockIdx.x * 4 + (threadIdx.x >> 5);
    if (tile >= (M >> 4) * ngroups) return;          // wave-uniform exit
    int mt = tile / ngroups, ng = tile % ngroups;
    int lane = threadIdx.x & 31;
    int grp  = lane >> 4;                            // 0: K 0..7/16..23, 1: 8..15/24..31
    int lr   = lane & 15;
    int kb   = grp * 8;

    const unsigned short* arow = A + (size_t)(mt * 16 + lr) * K + kb;
    const unsigned short* br0  = Bw + (size_t)(ng * 64 +  0 + lr) * K + kb;
    const unsigned short* br1  = Bw + (size_t)(ng * 64 + 16 + lr) * K + kb;
    const unsigned short* br2  = Bw + (size_t)(ng * 64 + 32 + lr) * K + kb;
    const unsigned short* br3  = Bw + (size_t)(ng * 64 + 48 + lr) * K + kb;
    v8f acc0 = {}, acc1 = {}, acc2 = {}, acc3 = {};
    for (int kt = 0; kt < K; kt += 32) {
        ABpack a, b0, b1, b2, b3;                    // all loads before any wmma
        a.q[0]  = *(const uint4*)(arow + kt);
        a.q[1]  = *(const uint4*)(arow + kt + 16);
        b0.q[0] = *(const uint4*)(br0 + kt);
        b0.q[1] = *(const uint4*)(br0 + kt + 16);
        b1.q[0] = *(const uint4*)(br1 + kt);
        b1.q[1] = *(const uint4*)(br1 + kt + 16);
        b2.q[0] = *(const uint4*)(br2 + kt);
        b2.q[1] = *(const uint4*)(br2 + kt + 16);
        b3.q[0] = *(const uint4*)(br3 + kt);
        b3.q[1] = *(const uint4*)(br3 + kt + 16);
        acc0 = __builtin_amdgcn_wmma_f32_16x16x32_bf16(
            false, a.v, false, b0.v, (short)0, acc0, false, false);
        acc1 = __builtin_amdgcn_wmma_f32_16x16x32_bf16(
            false, a.v, false, b1.v, (short)0, acc1, false, false);
        acc2 = __builtin_amdgcn_wmma_f32_16x16x32_bf16(
            false, a.v, false, b2.v, (short)0, acc2, false, false);
        acc3 = __builtin_amdgcn_wmma_f32_16x16x32_bf16(
            false, a.v, false, b3.v, (short)0, acc3, false, false);
    }
    v8f accs[4] = { acc0, acc1, acc2, acc3 };
#pragma unroll
    for (int j = 0; j < 4; ++j) {
#pragma unroll
        for (int r = 0; r < 8; ++r) {
            int m = mt * 16 + r + 8 * grp;           // C layout: vgpr r -> M = r + 8*grp
            int n = ng * 64 + j * 16 + lr;
            float v = accs[j][r];
            if (bias)  v += bias[n];
            if (resid) v += resid[(size_t)m * N + n];
            if (relu)  v = fmaxf(v, 0.f);
            if (Cf) Cf[(size_t)m * N + n] = v;
            if (Cb) Cb[(size_t)m * N + n] = f2bf(v);
        }
    }
}

// --- edge kernel: v1 = relu(F1[j] + Wx@(dx - h_i)); v2 = relu(v1@W1^T+b1);
//     v3 = v2@W2^T+b2; atomic-max scatter into AGG[i]. 64 edges/block. ------
__global__ __launch_bounds__(128)
void edge_k(const int* __restrict__ eidx, const float* __restrict__ xyz,
            const float* __restrict__ Hf, const float* __restrict__ F1,
            const float* __restrict__ fW0 /*[128,259] f32*/,
            const unsigned short* __restrict__ fW1b, const float* __restrict__ fb1,
            const unsigned short* __restrict__ fW2b, const float* __restrict__ fb2,
            float* __restrict__ AGG) {
    __shared__ __align__(16) unsigned short v1[EPB][128];
    __shared__ __align__(16) unsigned short v2[EPB][64];
    __shared__ float wx[128][3];
    __shared__ float wv3[EPB][3];
    __shared__ int   ii[EPB];
    __shared__ int   jj[EPB];

    int t  = threadIdx.x;
    int e0 = blockIdx.x * EPB;

    if (t < 128) {                                   // stage Wx = fW0[:,0:3]
        wx[t][0] = fW0[t * 259 + 0];
        wx[t][1] = fW0[t * 259 + 1];
        wx[t][2] = fW0[t * 259 + 2];
    }
    if (t < EPB) {                                   // per-edge geometry
        int j = eidx[e0 + t];
        int i = eidx[NE + e0 + t];
        jj[t] = j; ii[t] = i;
        wv3[t][0] = xyz[j * 3 + 0] - xyz[i * 3 + 0] - Hf[i * 3 + 0];
        wv3[t][1] = xyz[j * 3 + 1] - xyz[i * 3 + 1] - Hf[i * 3 + 1];
        wv3[t][2] = xyz[j * 3 + 2] - xyz[i * 3 + 2] - Hf[i * 3 + 2];
    }
    __syncthreads();

    {   // v1: gather F1 row + rank-3 correction + ReLU -> LDS bf16
        int e = t >> 1, c0 = (t & 1) * 64;
        int j = jj[e];
        float w0 = wv3[e][0], w1 = wv3[e][1], w2 = wv3[e][2];
        const float* frow = F1 + (size_t)j * 128 + c0;
#pragma unroll 8
        for (int c = 0; c < 64; ++c) {
            float v = frow[c] + wx[c0 + c][0] * w0 + wx[c0 + c][1] * w1
                              + wx[c0 + c][2] * w2;
            v1[e][c0 + c] = f2bf(fmaxf(v, 0.f));
        }
    }
    __syncthreads();

    int wvid = t >> 5, lane = t & 31;
    int grp = lane >> 4, lr = lane & 15, kb = grp * 8;

    // GEMM1: [64,128] x [128,64] -> v2 (relu); one A load feeds 4 WMMAs
    {
        const unsigned short* br0 = fW1b + (size_t)( 0 + lr) * 128 + kb;
        const unsigned short* br1 = fW1b + (size_t)(16 + lr) * 128 + kb;
        const unsigned short* br2 = fW1b + (size_t)(32 + lr) * 128 + kb;
        const unsigned short* br3 = fW1b + (size_t)(48 + lr) * 128 + kb;
        v8f acc0 = {}, acc1 = {}, acc2 = {}, acc3 = {};
#pragma unroll
        for (int kt = 0; kt < 128; kt += 32) {
            ABpack a, b0, b1, b2, b3;
            a.q[0]  = *(const uint4*)&v1[wvid * 16 + lr][kt + kb];
            a.q[1]  = *(const uint4*)&v1[wvid * 16 + lr][kt + kb + 16];
            b0.q[0] = *(const uint4*)(br0 + kt);
            b0.q[1] = *(const uint4*)(br0 + kt + 16);
            b1.q[0] = *(const uint4*)(br1 + kt);
            b1.q[1] = *(const uint4*)(br1 + kt + 16);
            b2.q[0] = *(const uint4*)(br2 + kt);
            b2.q[1] = *(const uint4*)(br2 + kt + 16);
            b3.q[0] = *(const uint4*)(br3 + kt);
            b3.q[1] = *(const uint4*)(br3 + kt + 16);
            acc0 = __builtin_amdgcn_wmma_f32_16x16x32_bf16(
                false, a.v, false, b0.v, (short)0, acc0, false, false);
            acc1 = __builtin_amdgcn_wmma_f32_16x16x32_bf16(
                false, a.v, false, b1.v, (short)0, acc1, false, false);
            acc2 = __builtin_amdgcn_wmma_f32_16x16x32_bf16(
                false, a.v, false, b2.v, (short)0, acc2, false, false);
            acc3 = __builtin_amdgcn_wmma_f32_16x16x32_bf16(
                false, a.v, false, b3.v, (short)0, acc3, false, false);
        }
        v8f accs[4] = { acc0, acc1, acc2, acc3 };
#pragma unroll
        for (int nt = 0; nt < 4; ++nt)
#pragma unroll
            for (int r = 0; r < 8; ++r) {
                int m = wvid * 16 + r + 8 * grp;
                int n = nt * 16 + lr;
                v2[m][n] = f2bf(fmaxf(accs[nt][r] + fb1[n], 0.f));
            }
    }
    __syncthreads();

    // GEMM2: [64,64] x [64,256] -> v3, scatter max into AGG.
    // 16 N-tiles processed as 4 groups of 4 sharing each A load.
#pragma unroll
    for (int gN = 0; gN < 4; ++gN) {
        const unsigned short* br0 = fW2b + (size_t)((gN * 4 + 0) * 16 + lr) * 64 + kb;
        const unsigned short* br1 = fW2b + (size_t)((gN * 4 + 1) * 16 + lr) * 64 + kb;
        const unsigned short* br2 = fW2b + (size_t)((gN * 4 + 2) * 16 + lr) * 64 + kb;
        const unsigned short* br3 = fW2b + (size_t)((gN * 4 + 3) * 16 + lr) * 64 + kb;
        v8f acc0 = {}, acc1 = {}, acc2 = {}, acc3 = {};
#pragma unroll
        for (int kt = 0; kt < 64; kt += 32) {
            ABpack a, b0, b1, b2, b3;
            a.q[0]  = *(const uint4*)&v2[wvid * 16 + lr][kt + kb];
            a.q[1]  = *(const uint4*)&v2[wvid * 16 + lr][kt + kb + 16];
            b0.q[0] = *(const uint4*)(br0 + kt);
            b0.q[1] = *(const uint4*)(br0 + kt + 16);
            b1.q[0] = *(const uint4*)(br1 + kt);
            b1.q[1] = *(const uint4*)(br1 + kt + 16);
            b2.q[0] = *(const uint4*)(br2 + kt);
            b2.q[1] = *(const uint4*)(br2 + kt + 16);
            b3.q[0] = *(const uint4*)(br3 + kt);
            b3.q[1] = *(const uint4*)(br3 + kt + 16);
            acc0 = __builtin_amdgcn_wmma_f32_16x16x32_bf16(
                false, a.v, false, b0.v, (short)0, acc0, false, false);
            acc1 = __builtin_amdgcn_wmma_f32_16x16x32_bf16(
                false, a.v, false, b1.v, (short)0, acc1, false, false);
            acc2 = __builtin_amdgcn_wmma_f32_16x16x32_bf16(
                false, a.v, false, b2.v, (short)0, acc2, false, false);
            acc3 = __builtin_amdgcn_wmma_f32_16x16x32_bf16(
                false, a.v, false, b3.v, (short)0, acc3, false, false);
        }
        v8f accs[4] = { acc0, acc1, acc2, acc3 };
#pragma unroll
        for (int j = 0; j < 4; ++j)
#pragma unroll
            for (int r = 0; r < 8; ++r) {
                int m = wvid * 16 + r + 8 * grp;
                int n = (gN * 4 + j) * 16 + lr;
                float val = accs[j][r] + fb2[n];
                atomic_max_f32(AGG + (size_t)ii[m] * 256 + n, val);
            }
    }
}

// --- InstanceNorm over feature dim (C=256) + ReLU, one row per wave --------
__global__ __launch_bounds__(256)
void inorm_relu_k(const float* __restrict__ X, unsigned short* __restrict__ Yb, int M) {
    int row  = blockIdx.x * 8 + (threadIdx.x >> 5);
    int lane = threadIdx.x & 31;
    if (row >= M) return;
    const float* x = X + (size_t)row * 256;
    float vals[8], s = 0.f, ss = 0.f;
#pragma unroll
    for (int i = 0; i < 8; ++i) {
        float v = x[lane + 32 * i];
        vals[i] = v; s += v; ss += v * v;
    }
#pragma unroll
    for (int off = 16; off; off >>= 1) {
        s  += __shfl_xor(s, off, 32);
        ss += __shfl_xor(ss, off, 32);
    }
    float mean = s * (1.f / 256.f);
    float var  = ss * (1.f / 256.f) - mean * mean;
    float rs   = rsqrtf(var + 1e-5f);
    unsigned short* y = Yb + (size_t)row * 256;
#pragma unroll
    for (int i = 0; i < 8; ++i)
        y[lane + 32 * i] = f2bf(fmaxf((vals[i] - mean) * rs, 0.f));
}

// --- tiny final layers (Nout in {1,3,7}): plain dot products ---------------
__global__ __launch_bounds__(256)
void small_out_k(const unsigned short* __restrict__ A, int K,
                 const float* __restrict__ W, const float* __restrict__ b,
                 int Nout, float* __restrict__ out, int M) {
    int t = blockIdx.x * 256 + threadIdx.x;
    if (t >= M * Nout) return;
    int m = t / Nout, n = t % Nout;
    const unsigned short* a = A + (size_t)m * K;
    const float* w = W + (size_t)n * K;
    float s = b[n];
    for (int k = 0; k < K; ++k) s += bf2f(a[k]) * w[k];
    out[(size_t)m * Nout + n] = s;
}

// ---------------------------------------------------------------------------
static inline dim3 g1(int total, int per) { return dim3((total + per - 1) / per); }

extern "C" void kernel_launch(void* const* d_in, const int* in_sizes, int n_in,
                              void* d_out, int out_size, void* d_ws, size_t ws_size,
                              hipStream_t stream) {
    (void)in_sizes; (void)n_in; (void)out_size; (void)ws_size;
    // Input layout (jax pytree flatten: dict keys sorted -> cls, layers, loc;
    // per-layer dict -> f, g, h; each MLP -> (W,b)*3):
    const float* pooled = (const float*)d_in[0];
    const float* xyz    = (const float*)d_in[1];
    const int*   eidx   = (const int*)d_in[2];
    const float *cW0 = (const float*)d_in[3], *cb0 = (const float*)d_in[4];
    const float *cW1 = (const float*)d_in[5], *cb1 = (const float*)d_in[6];
    const float *cW2 = (const float*)d_in[7], *cb2 = (const float*)d_in[8];
    const int L2 = 9 + 2 * 18;   // layers[2] base (only live layer)
    const float *fW0 = (const float*)d_in[L2 + 0],  *fb0 = (const float*)d_in[L2 + 1];
    const float *fW1 = (const float*)d_in[L2 + 2],  *fb1 = (const float*)d_in[L2 + 3];
    const float *fW2 = (const float*)d_in[L2 + 4],  *fb2 = (const float*)d_in[L2 + 5];
    const float *gW0 = (const float*)d_in[L2 + 6],  *gb0 = (const float*)d_in[L2 + 7];
    const float *gW1 = (const float*)d_in[L2 + 8],  *gb1 = (const float*)d_in[L2 + 9];
    const float *gW2 = (const float*)d_in[L2 + 10], *gb2 = (const float*)d_in[L2 + 11];
    const float *hW0 = (const float*)d_in[L2 + 12], *hb0 = (const float*)d_in[L2 + 13];
    const float *hW1 = (const float*)d_in[L2 + 14], *hb1 = (const float*)d_in[L2 + 15];
    const float *hW2 = (const float*)d_in[L2 + 16], *hb2 = (const float*)d_in[L2 + 17];
    const float *lW0 = (const float*)d_in[63], *lb0 = (const float*)d_in[64];
    const float *lW1 = (const float*)d_in[65], *lb1 = (const float*)d_in[66];
    const float *lW2 = (const float*)d_in[67], *lb2 = (const float*)d_in[68];
    float* out_cls = (float*)d_out;
    float* out_loc = (float*)d_out + BN;

    // Workspace bump allocator (256B aligned)
    char* ws = (char*)d_ws; size_t off = 0;
    auto alloc = [&](size_t bytes) -> char* {
        char* p = ws + off; off = (off + bytes + 255) & ~(size_t)255; return p;
    };
    unsigned short* Sb   = (unsigned short*)alloc((size_t)BN * SD * 2);
    unsigned short* fWsb = (unsigned short*)alloc(128 * 256 * 2);
    unsigned short* fW1b = (unsigned short*)alloc(64 * 128 * 2);
    unsigned short* fW2b = (unsigned short*)alloc(256 * 64 * 2);
    unsigned short* gW0b = (unsigned short*)alloc(128 * 256 * 2);
    unsigned short* gW1b = (unsigned short*)alloc(64 * 128 * 2);
    unsigned short* gW2b = (unsigned short*)alloc(256 * 64 * 2);
    unsigned short* hW0b = (unsigned short*)alloc(128 * 256 * 2);
    unsigned short* hW1b = (unsigned short*)alloc(64 * 128 * 2);
    unsigned short* cW0b = (unsigned short*)alloc(256 * 256 * 2);
    unsigned short* cW1b = (unsigned short*)alloc(256 * 256 * 2);
    unsigned short* lW0b = (unsigned short*)alloc(256 * 256 * 2);
    unsigned short* lW1b = (unsigned short*)alloc(256 * 256 * 2);
    float*          F1   = (float*)alloc((size_t)BN * 128 * 4);
    unsigned short* M1b  = (unsigned short*)alloc((size_t)BN * 128 * 2); // h1/g1 tmp
    unsigned short* M2b  = (unsigned short*)alloc((size_t)BN * 64 * 2);  // h2/g2 tmp
    float*          Hf   = (float*)alloc((size_t)BN * 3 * 4);
    float*          AGG  = (float*)alloc((size_t)BN * 256 * 4);
    unsigned short* AGGb = (unsigned short*)alloc((size_t)BN * 256 * 2);
    unsigned short* OUTb = (unsigned short*)alloc((size_t)BN * 256 * 2);
    float*          Tf   = (float*)alloc((size_t)BN * 256 * 4);
    unsigned short* Tb   = (unsigned short*)alloc((size_t)BN * 256 * 2);

    // 1) bf16 conversions: activations + all WMMA-consumed weights
    cvt_k<<<g1(BN * SD, 256), 256, 0, stream>>>(pooled, 256, 0, Sb, BN, 256);
    cvt_k<<<g1(128 * 256, 256), 256, 0, stream>>>(fW0, 259, 3, fWsb, 128, 256); // W[:,3:]
    cvt_k<<<g1(64 * 128, 256), 256, 0, stream>>>(fW1, 128, 0, fW1b, 64, 128);
    cvt_k<<<g1(256 * 64, 256), 256, 0, stream>>>(fW2, 64, 0, fW2b, 256, 64);
    cvt_k<<<g1(128 * 256, 256), 256, 0, stream>>>(gW0, 256, 0, gW0b, 128, 256);
    cvt_k<<<g1(64 * 128, 256), 256, 0, stream>>>(gW1, 128, 0, gW1b, 64, 128);
    cvt_k<<<g1(256 * 64, 256), 256, 0, stream>>>(gW2, 64, 0, gW2b, 256, 64);
    cvt_k<<<g1(128 * 256, 256), 256, 0, stream>>>(hW0, 256, 0, hW0b, 128, 256);
    cvt_k<<<g1(64 * 128, 256), 256, 0, stream>>>(hW1, 128, 0, hW1b, 64, 128);
    cvt_k<<<g1(256 * 256, 256), 256, 0, stream>>>(cW0, 256, 0, cW0b, 256, 256);
    cvt_k<<<g1(256 * 256, 256), 256, 0, stream>>>(cW1, 256, 0, cW1b, 256, 256);
    cvt_k<<<g1(256 * 256, 256), 256, 0, stream>>>(lW0, 256, 0, lW0b, 256, 256);
    cvt_k<<<g1(256 * 256, 256), 256, 0, stream>>>(lW1, 256, 0, lW1b, 256, 256);
    fill_neginf_k<<<g1(BN * 256, 256), 256, 0, stream>>>(AGG, BN * 256);

    // 2) node-level precompute: F1 = S @ fW0[:,3:]^T + fb0 ; H = h_mlp(S)
    gemm_bf16_k<<<g1((BN/16)*(128/64), 4), 128, 0, stream>>>(
        Sb, fWsb, fb0, nullptr, F1, nullptr, BN, 128, 256, 0);
    gemm_bf16_k<<<g1((BN/16)*(128/64), 4), 128, 0, stream>>>(
        Sb, hW0b, hb0, nullptr, nullptr, M1b, BN, 128, 256, 1);
    gemm_bf16_k<<<g1((BN/16)*(64/64), 4), 128, 0, stream>>>(
        M1b, hW1b, hb1, nullptr, nullptr, M2b, BN, 64, 128, 1);
    small_out_k<<<g1(BN * 3, 256), 256, 0, stream>>>(M2b, 64, hW2, hb2, 3, Hf, BN);

    // 3) per-edge message MLP (WMMA from LDS) + atomic-max segment scatter
    edge_k<<<dim3(NE / EPB), 128, 0, stream>>>(
        eidx, xyz, Hf, F1, fW0, fW1b, fb1, fW2b, fb2, AGG);

    // 4) empty-segment fixup + g-MLP with residual: OUT = S + g(agg)
    fix_agg_k<<<g1(BN * 256, 256), 256, 0, stream>>>(AGG, AGGb, BN * 256);
    gemm_bf16_k<<<g1((BN/16)*(128/64), 4), 128, 0, stream>>>(
        AGGb, gW0b, gb0, nullptr, nullptr, M1b, BN, 128, 256, 1);
    gemm_bf16_k<<<g1((BN/16)*(64/64), 4), 128, 0, stream>>>(
        M1b, gW1b, gb1, nullptr, nullptr, M2b, BN, 64, 128, 1);
    gemm_bf16_k<<<g1((BN/16)*(256/64), 4), 128, 0, stream>>>(
        M2b, gW2b, gb2, pooled, nullptr, OUTb, BN, 256, 64, 0);

    // 5) cls head: relu(inorm(x@W^T+b)) x2, then 256->1
    gemm_bf16_k<<<g1((BN/16)*(256/64), 4), 128, 0, stream>>>(
        OUTb, cW0b, cb0, nullptr, Tf, nullptr, BN, 256, 256, 0);
    inorm_relu_k<<<g1(BN, 8), 256, 0, stream>>>(Tf, Tb, BN);
    gemm_bf16_k<<<g1((BN/16)*(256/64), 4), 128, 0, stream>>>(
        Tb, cW1b, cb1, nullptr, Tf, nullptr, BN, 256, 256, 0);
    inorm_relu_k<<<g1(BN, 8), 256, 0, stream>>>(Tf, Tb, BN);
    small_out_k<<<g1(BN * 1, 256), 256, 0, stream>>>(Tb, 256, cW2, cb2, 1, out_cls, BN);

    // 6) loc head: same, 256->7
    gemm_bf16_k<<<g1((BN/16)*(256/64), 4), 128, 0, stream>>>(
        OUTb, lW0b, lb0, nullptr, Tf, nullptr, BN, 256, 256, 0);
    inorm_relu_k<<<g1(BN, 8), 256, 0, stream>>>(Tf, Tb, BN);
    gemm_bf16_k<<<g1((BN/16)*(256/64), 4), 128, 0, stream>>>(
        Tb, lW1b, lb1, nullptr, Tf, nullptr, BN, 256, 256, 0);
    inorm_relu_k<<<g1(BN, 8), 256, 0, stream>>>(Tf, Tb, BN);
    small_out_k<<<g1(BN * 7, 256), 256, 0, stream>>>(Tb, 256, lW2, lb2, 7, out_loc, BN);
}
